// MultiHeadAttention_76407468196286
// MI455X (gfx1250) — matrix operand order repaired and checked by reading, not compiled, not run on hardware
//
#include <hip/hip_runtime.h>
#include <hip/hip_bf16.h>

typedef __attribute__((ext_vector_type(16))) _Float16 v16h;
typedef __attribute__((ext_vector_type(8)))  float    v8f;

#define WMMA_F16(A, Bm, C) \
  __builtin_amdgcn_wmma_f32_16x16x32_f16(false, (A), false, (Bm), (short)0, (C), false, false)

// Problem constants
static constexpr int Bc = 2;
static constexpr int Tc = 2048;
static constexpr int Mc = 2048;
static constexpr int Dc = 128;
static constexpr int Hc = 16;

// ---------------------------------------------------------------------------
// Kernel 1: QKV projection  [B*T, M] x [M, H*D] -> q/k/v with RoPE + muP scale
// Block tile 128x128 (one head per col-tile), BK=32, 8 waves (2x4), each wave
// computes 64x32 via 4x2 wmma_f32_16x16x32_f16 accumulators.
// Weight tile staged TRANSPOSED in LDS so B-fragments are 2x ds_load_b128.
// ---------------------------------------------------------------------------
__global__ __launch_bounds__(256)
void qkv_kernel(const float* __restrict__ x,
                const float* __restrict__ wq,
                const float* __restrict__ wk,
                const float* __restrict__ wv,
                _Float16* __restrict__ Qh,
                _Float16* __restrict__ Kh,
                _Float16* __restrict__ Vh)
{
    __shared__ __align__(16) _Float16 sA [128][40];   // [row][k]
    __shared__ __align__(16) _Float16 sBt[128][40];   // [col][k]  (transposed)
    __shared__ __align__(16) _Float16 sC [128][136];

    const int tid  = threadIdx.x;
    const int wave = tid >> 5, lane = tid & 31;
    const int lh = lane >> 4, ln = lane & 15;
    const int wm = wave >> 2, wn = wave & 3;      // 2x4 wave grid
    const int row0 = blockIdx.x * 128;            // rows of B*T
    const int h    = blockIdx.y;                  // head == 128-col tile
    const int z    = blockIdx.z;                  // 0=q 1=k 2=v
    const float* __restrict__ w = (z == 0) ? wq : (z == 1) ? wk : wv;

    v8f acc[4][2];
    #pragma unroll
    for (int tm = 0; tm < 4; tm++)
        #pragma unroll
        for (int tn = 0; tn < 2; tn++)
            acc[tm][tn] = (v8f){0.f,0.f,0.f,0.f,0.f,0.f,0.f,0.f};

    for (int k0 = 0; k0 < Mc; k0 += 32) {
        __syncthreads();
        { // A tile 128x32 f32 -> f16: 2 threads per row, 16 contiguous elems
            const int r  = tid >> 1;
            const int c0 = (tid & 1) * 16;
            const float* src = x + (size_t)(row0 + r) * Mc + k0 + c0;
            #pragma unroll
            for (int i = 0; i < 16; i++) sA[r][c0 + i] = (_Float16)src[i];
        }
        { // B tile 32x128 f32 -> f16, stored transposed [col][k].
          // col = tid&127 keeps global reads lane-coalesced; LDS stores are
          // 16 contiguous halves -> vector ds stores.
            const int col = tid & 127;
            const int kb  = (tid >> 7) * 16;       // 0 or 16
            const float* src = w + (size_t)(k0 + kb) * Mc + h * Dc + col;
            #pragma unroll
            for (int i = 0; i < 16; i++)
                sBt[col][kb + i] = (_Float16)src[(size_t)i * Mc];
        }
        __syncthreads();

        v16h bfr[2];
        #pragma unroll
        for (int tn = 0; tn < 2; tn++) {
            const int col = wn * 32 + tn * 16 + ln;     // n = lane%16
            #pragma unroll
            for (int idx = 0; idx < 16; idx++)          // B: k = lh*16 + idx (contiguous)
                bfr[tn][idx] = sBt[col][lh * 16 + idx];
        }
        #pragma unroll
        for (int tm = 0; tm < 4; tm++) {
            v16h a;
            const int rr = wm * 64 + tm * 16 + ln;      // m = lane%16
            #pragma unroll
            for (int idx = 0; idx < 16; idx++) {        // A: k = lh*16 + idx (contiguous)
                const int kk = lh * 16 + idx;
                a[idx] = sA[rr][kk];
            }
            #pragma unroll
            for (int tn = 0; tn < 2; tn++)
                acc[tm][tn] = WMMA_F16(a, bfr[tn], acc[tm][tn]);
        }
    }

    // Stage tile as f16 in LDS for the RoPE epilogue (needs cols d and d+64).
    __syncthreads();
    #pragma unroll
    for (int tm = 0; tm < 4; tm++)
        #pragma unroll
        for (int tn = 0; tn < 2; tn++) {
            const int cc = wn * 32 + tn * 16 + ln;
            #pragma unroll
            for (int j = 0; j < 8; j++)                 // C: m = j + 8*lh
                sC[wm * 64 + tm * 16 + j + 8 * lh][cc] = (_Float16)acc[tm][tn][j];
        }
    __syncthreads();

    const bool rope  = (z < 2);
    const float mult = rope ? 0.08838834764831845f : 1.0f;   // 128^-0.5
    _Float16* __restrict__ out = (z == 0) ? Qh : (z == 1) ? Kh : Vh;

    const int r    = tid >> 1;            // local row 0..127
    const int half = tid & 1;             // which 32 of d in [0,64)
    const int grow = row0 + r;
    const int b = grow / Tc;
    const int t = grow % Tc;
    const size_t obase = (((size_t)b * Hc + h) * Tc + t) * Dc;

    #pragma unroll
    for (int dd = 0; dd < 32; dd++) {
        const int d = half * 32 + dd;                    // 0..63
        const float e = (float)sC[r][d];
        const float o = (float)sC[r][d + 64];
        float oe, oo;
        if (rope) {
            const float ang = powf(10000.0f, -(float)d * (1.0f / 64.0f));
            const float rad = (float)t * ang;
            const float cs = cosf(rad), sn = sinf(rad);
            oe = (e * cs - o * sn) * mult;
            oo = (e * sn + o * cs) * mult;
        } else { oe = e; oo = o; }
        out[obase + d]      = (_Float16)oe;
        out[obase + d + 64] = (_Float16)oo;
    }
}

// ---------------------------------------------------------------------------
// Kernel 2: causal flash attention per (b, h, 128-query tile).
// 8 waves x 16 query rows; key tiles of 64; K staged row-major (S-fragments
// read contiguous d), V staged TRANSPOSED [d][key] (PV-fragments read
// contiguous key) -> all WMMA operand reads are vector ds loads.
// ---------------------------------------------------------------------------
__global__ __launch_bounds__(256)
void attn_kernel(const _Float16* __restrict__ Qh,
                 const _Float16* __restrict__ Kh,
                 const _Float16* __restrict__ Vh,
                 _Float16* __restrict__ Oh)
{
    __shared__ __align__(16) _Float16 sK [64][136];     // [key][d]
    __shared__ __align__(16) _Float16 sVt[128][72];     // [d][key] (transposed)
    __shared__ __align__(16) _Float16 sP [8][16][72];

    const int tid  = threadIdx.x;
    const int wave = tid >> 5, lane = tid & 31;
    const int lh = lane >> 4, ln = lane & 15;
    const int qt = blockIdx.x, h = blockIdx.y, b = blockIdx.z;
    const size_t hb = ((size_t)b * Hc + h) * (size_t)Tc * Dc;  // [T,D] plane
    const int qbase = qt * 128 + wave * 16;

    // Q fragments for this wave's 16 rows (A layout), kept in registers.
    v16h qf[4];
    #pragma unroll
    for (int tq = 0; tq < 4; tq++) {
        const _Float16* qp = Qh + hb + (size_t)(qbase + ln) * Dc + tq * 32;
        #pragma unroll
        for (int idx = 0; idx < 16; idx++)
            qf[tq][idx] = qp[lh * 16 + idx];
    }

    v8f o[8];
    #pragma unroll
    for (int dt = 0; dt < 8; dt++)
        o[dt] = (v8f){0.f,0.f,0.f,0.f,0.f,0.f,0.f,0.f};
    float rm[8], rl[8];
    #pragma unroll
    for (int j = 0; j < 8; j++) { rm[j] = -INFINITY; rl[j] = 0.f; }

    const int nkt = qt * 2 + 2;                  // key tiles 0 .. qt*128+127
    for (int kt = 0; kt < nkt; kt++) {
        __syncthreads();
        {   // K tile 64x128 halves row-major: 1024 uint4 vector loads/stores
            const uint4* ksrc = (const uint4*)(Kh + hb + (size_t)kt * 64 * Dc);
            #pragma unroll
            for (int i = 0; i < 4; i++) {
                const int lin = tid + i * 256;
                const int r = lin >> 4, cv = lin & 15;
                *(uint4*)&sK[r][cv * 8] = ksrc[lin];
            }
        }
        {   // V tile staged transposed: d = tid&127 (lane-coalesced global
            // u16 reads), 32 contiguous key-halves per thread in LDS.
            const int d   = tid & 127;
            const int kb  = (tid >> 7) * 32;     // 0 or 32
            const _Float16* src = Vh + hb + (size_t)(kt * 64 + kb) * Dc + d;
            #pragma unroll
            for (int i = 0; i < 32; i++)
                sVt[d][kb + i] = src[(size_t)i * Dc];
        }
        if (kt + 1 < nkt) {   // prefetch next tiles (global_prefetch_b8)
            const char* nk = (const char*)(Kh + hb + (size_t)(kt + 1) * 64 * Dc);
            const char* nv = (const char*)(Vh + hb + (size_t)(kt + 1) * 64 * Dc);
            __builtin_prefetch(nk + tid * 64, 0, 1);
            __builtin_prefetch(nv + tid * 64, 0, 1);
        }
        __syncthreads();
        if (kt * 64 > qbase + 15) continue;      // fully future for this wave

        // S = Q K^T : 16 x 64, four 16x16 accumulators
        v8f s[4];
        #pragma unroll
        for (int tn = 0; tn < 4; tn++) {
            v8f a = (v8f){0.f,0.f,0.f,0.f,0.f,0.f,0.f,0.f};
            #pragma unroll
            for (int tq = 0; tq < 4; tq++) {
                v16h bf;
                #pragma unroll
                for (int idx = 0; idx < 16; idx++)   // B: k(d) = tq*32 + lh*16 + idx (contig)
                    bf[idx] = sK[tn * 16 + ln][tq * 32 + lh * 16 + idx];
                a = WMMA_F16(qf[tq], bf, a);
            }
            s[tn] = a;
        }

        // causal mask on diagonal tiles
        if (kt * 64 + 63 > qbase) {
            #pragma unroll
            for (int tn = 0; tn < 4; tn++) {
                const int key = kt * 64 + tn * 16 + ln;
                #pragma unroll
                for (int j = 0; j < 8; j++)
                    if (key > qbase + j + 8 * lh) s[tn][j] -= 1e30f;
            }
        }

        // online softmax (row stats shared by the 16 lanes holding each row)
        #pragma unroll
        for (int j = 0; j < 8; j++) {
            float m = fmaxf(fmaxf(s[0][j], s[1][j]), fmaxf(s[2][j], s[3][j]));
            m = fmaxf(m, __shfl_xor(m, 1, 32));
            m = fmaxf(m, __shfl_xor(m, 2, 32));
            m = fmaxf(m, __shfl_xor(m, 4, 32));
            m = fmaxf(m, __shfl_xor(m, 8, 32));
            const float mnew  = fmaxf(rm[j], m);
            const float alpha = __expf(rm[j] - mnew);
            rm[j] = mnew;
            float rs = 0.f;
            #pragma unroll
            for (int tn = 0; tn < 4; tn++) {
                const float p = __expf(s[tn][j] - mnew);
                s[tn][j] = p;
                rs += p;
            }
            rs += __shfl_xor(rs, 1, 32);
            rs += __shfl_xor(rs, 2, 32);
            rs += __shfl_xor(rs, 4, 32);
            rs += __shfl_xor(rs, 8, 32);
            rl[j] = rl[j] * alpha + rs;
            #pragma unroll
            for (int dt = 0; dt < 8; dt++) o[dt][j] *= alpha;
        }

        // stage P (C layout -> A layout) through per-wave LDS patch
        #pragma unroll
        for (int tn = 0; tn < 4; tn++)
            #pragma unroll
            for (int j = 0; j < 8; j++)
                sP[wave][j + 8 * lh][tn * 16 + ln] = (_Float16)s[tn][j];
        asm volatile("s_wait_dscnt 0" ::: "memory");

        v16h pa[2];
        #pragma unroll
        for (int th = 0; th < 2; th++)
            #pragma unroll
            for (int idx = 0; idx < 16; idx++)
                pa[th][idx] = sP[wave][ln][th * 32 + lh * 16 + idx];

        // O += P x V  (V fragments: contiguous key reads from sVt)
        #pragma unroll
        for (int dt = 0; dt < 8; dt++)
            #pragma unroll
            for (int th = 0; th < 2; th++) {
                v16h bf;
                #pragma unroll
                for (int idx = 0; idx < 16; idx++)   // B: k(key) = th*32 + lh*16 + idx (contig)
                    bf[idx] = sVt[dt * 16 + ln][th * 32 + lh * 16 + idx];
                o[dt] = WMMA_F16(pa[th], bf, o[dt]);
            }
    }

    // normalize + store O in [B,T,H,D] f16 for the output projection
    #pragma unroll
    for (int j = 0; j < 8; j++) {
        const float inv = 1.0f / rl[j];
        const int t = qbase + j + 8 * lh;
        const size_t ob = (((size_t)b * Tc + t) * Hc + h) * Dc;
        #pragma unroll
        for (int dt = 0; dt < 8; dt++)
            Oh[ob + dt * 16 + ln] = (_Float16)(o[dt][j] * inv);
    }
}

// ---------------------------------------------------------------------------
// Kernel 3: output projection  [B*T, H*D](f16) x [H*D, M](f32->f16) -> f32 out
// ---------------------------------------------------------------------------
__global__ __launch_bounds__(256)
void proj_kernel(const _Float16* __restrict__ Oh,
                 const float* __restrict__ wo,
                 float* __restrict__ out)
{
    __shared__ __align__(16) _Float16 sA [128][40];
    __shared__ __align__(16) _Float16 sBt[128][40];    // [col][k] (transposed)

    const int tid  = threadIdx.x;
    const int wave = tid >> 5, lane = tid & 31;
    const int lh = lane >> 4, ln = lane & 15;
    const int wm = wave >> 2, wn = wave & 3;
    const int row0 = blockIdx.x * 128;
    const int col0 = blockIdx.y * 128;

    v8f acc[4][2];
    #pragma unroll
    for (int tm = 0; tm < 4; tm++)
        #pragma unroll
        for (int tn = 0; tn < 2; tn++)
            acc[tm][tn] = (v8f){0.f,0.f,0.f,0.f,0.f,0.f,0.f,0.f};

    for (int k0 = 0; k0 < Mc; k0 += 32) {
        __syncthreads();
        {
            const int r  = tid >> 1;
            const int c0 = (tid & 1) * 16;
            const _Float16* src = Oh + (size_t)(row0 + r) * Mc + k0 + c0;
            #pragma unroll
            for (int i = 0; i < 16; i++) sA[r][c0 + i] = src[i];
        }
        {
            const int col = tid & 127;
            const int kb  = (tid >> 7) * 16;
            const float* src = wo + (size_t)(k0 + kb) * Mc + col0 + col;
            #pragma unroll
            for (int i = 0; i < 16; i++)
                sBt[col][kb + i] = (_Float16)src[(size_t)i * Mc];
        }
        __syncthreads();

        v16h bfr[2];
        #pragma unroll
        for (int tn = 0; tn < 2; tn++) {
            const int col = wn * 32 + tn * 16 + ln;
            #pragma unroll
            for (int idx = 0; idx < 16; idx++)
                bfr[tn][idx] = sBt[col][lh * 16 + idx];
        }
        #pragma unroll
        for (int tm = 0; tm < 4; tm++) {
            v16h a;
            const int rr = wm * 64 + tm * 16 + ln;
            #pragma unroll
            for (int idx = 0; idx < 16; idx++)
                a[idx] = sA[rr][lh * 16 + idx];
            #pragma unroll
            for (int tn = 0; tn < 2; tn++)
                acc[tm][tn] = WMMA_F16(a, bfr[tn], acc[tm][tn]);
        }
    }

    #pragma unroll
    for (int tm = 0; tm < 4; tm++)
        #pragma unroll
        for (int tn = 0; tn < 2; tn++) {
            const int gc = col0 + wn * 32 + tn * 16 + ln;
            #pragma unroll
            for (int j = 0; j < 8; j++) {
                const int gr = row0 + wm * 64 + tm * 16 + j + 8 * lh;
                out[(size_t)gr * Mc + gc] = acc[tm][tn][j];
            }
        }
}

// ---------------------------------------------------------------------------
extern "C" void kernel_launch(void* const* d_in, const int* in_sizes, int n_in,
                              void* d_out, int out_size, void* d_ws, size_t ws_size,
                              hipStream_t stream)
{
    (void)in_sizes; (void)n_in; (void)out_size; (void)ws_size;
    const float* x  = (const float*)d_in[0];
    const float* wq = (const float*)d_in[1];
    const float* wk = (const float*)d_in[2];
    const float* wv = (const float*)d_in[3];
    const float* wo = (const float*)d_in[4];
    float* out = (float*)d_out;

    const size_t N = (size_t)Bc * Hc * Tc * Dc;   // 8,388,608 per plane
    _Float16* Qh = (_Float16*)d_ws;
    _Float16* Kh = Qh + N;
    _Float16* Vh = Kh + N;
    _Float16* Oh = Vh + N;                        // [B,T,H,D]

    dim3 blk(256);
    qkv_kernel <<<dim3(Bc * Tc / 128, Hc, 3), blk, 0, stream>>>(x, wq, wk, wv, Qh, Kh, Vh);
    attn_kernel<<<dim3(Tc / 128, Hc, Bc),     blk, 0, stream>>>(Qh, Kh, Vh, Oh);
    proj_kernel<<<dim3(Bc * Tc / 128, Mc / 128), blk, 0, stream>>>(Oh, wo, out);
}
